// MultiHeadAttention_55877524521603
// MI455X (gfx1250) — matrix operand order repaired and checked by reading, not compiled, not run on hardware
//
#include <hip/hip_runtime.h>
#include <hip/hip_bf16.h>

typedef __attribute__((ext_vector_type(16))) _Float16 v16h;
typedef __attribute__((ext_vector_type(8)))  _Float16 v8h;
typedef __attribute__((ext_vector_type(8)))  float    v8f;

#define T_ 2048
#define B_ 2
#define E_ 1024
#define H_ 16
#define D_ 64
#define N_ (T_*B_)   // 4096 rows in the flattened [T*B, E] matrices

// ---------------------------------------------------------------------------
// WMMA helpers (CDNA5 gfx1250, wave32)
// ---------------------------------------------------------------------------
static __device__ inline v8f wmma16(v16h a, v16h b, v8f c) {
  // D = A(16x32 f16) * B(32x16 f16) + C(16x16 f32)
  return __builtin_amdgcn_wmma_f32_16x16x32_f16(
      /*neg_a=*/false, a, /*neg_b=*/false, b,
      /*c_mod=*/(short)0, c, /*reuse_a=*/false, /*reuse_b=*/false);
}

// A-fragment: 16x32 (MxK) f16. Lane L<16: M=L, K = {k0..k0+7, k0+16..k0+23};
// lane L>=16: M=L-16, K = {k0+8..k0+15, k0+24..k0+31}. Rows contiguous in K.
static __device__ inline v16h load_a_frag_g(const _Float16* __restrict__ base,
                                            int ld, int row0, int k0, int lane) {
  int row = row0 + (lane & 15);
  int kk  = k0 + ((lane & 16) ? 8 : 0);
  const _Float16* p = base + (size_t)row * (size_t)ld + kk;
  v8h lo = *(const v8h*)p;          // K = kk .. kk+7
  v8h hi = *(const v8h*)(p + 16);   // K = kk+16 .. kk+23
  v16h r;
#pragma unroll
  for (int i = 0; i < 8; ++i) { r[i] = lo[i]; r[8 + i] = hi[i]; }
  return r;
}

// B-fragment: 32x16 (KxN) f16 from a [N][K] (K-contiguous) matrix.
// Lane L: N = col0 + (L&15), K = {kk..kk+7, kk+16..kk+23}, kk = k0 + (L>=16 ? 8 : 0).
static __device__ inline v16h load_b_frag_g(const _Float16* __restrict__ base,
                                            int ld, int col0, int k0, int lane) {
  int col = col0 + (lane & 15);
  int kk  = k0 + ((lane & 16) ? 8 : 0);
  const _Float16* p = base + (size_t)col * (size_t)ld + kk;
  v8h lo = *(const v8h*)p;
  v8h hi = *(const v8h*)(p + 16);
  v16h r;
#pragma unroll
  for (int i = 0; i < 8; ++i) { r[i] = lo[i]; r[8 + i] = hi[i]; }
  return r;
}

// A-fragment from a 16x32 f16 LDS tile (row stride 32 halves).
static __device__ inline v16h load_a_frag_lds(const _Float16* tile, int lane) {
  int row = lane & 15;
  int kk  = (lane & 16) ? 8 : 0;
  const _Float16* p = tile + row * 32 + kk;
  v8h lo  = *(const v8h*)p;
  v8h hi2 = *(const v8h*)(p + 16);
  v16h r;
#pragma unroll
  for (int i = 0; i < 8; ++i) { r[i] = lo[i]; r[8 + i] = hi2[i]; }
  return r;
}

// Online-softmax update for one 16-row M-tile given two 16x16 score C-frags.
// Writes P (f16) into a 16x32 LDS tile for C->A re-layout.
static __device__ inline void softmax_update(
    v8f& sc0, v8f& sc1, float mrow[8], float lrow[8],
    v8f& o0, v8f& o1, v8f& o2, v8f& o3,
    _Float16* ptile, int lane) {
  const float L2E = 1.44269504088896340736f;
  int hi  = (lane & 16) ? 8 : 0;
  int col = lane & 15;
#pragma unroll
  for (int r = 0; r < 8; ++r) {
    float mx = fmaxf(sc0[r], sc1[r]);
#pragma unroll
    for (int off = 8; off >= 1; off >>= 1)
      mx = fmaxf(mx, __shfl_xor(mx, off, 32));
    float mnew = fmaxf(mrow[r], mx);
    float corr = exp2f((mrow[r] - mnew) * L2E);
    float p0 = exp2f((sc0[r] - mnew) * L2E);
    float p1 = exp2f((sc1[r] - mnew) * L2E);
    float rs = p0 + p1;
#pragma unroll
    for (int off = 8; off >= 1; off >>= 1)
      rs += __shfl_xor(rs, off, 32);
    lrow[r] = lrow[r] * corr + rs;
    mrow[r] = mnew;
    o0[r] *= corr; o1[r] *= corr; o2[r] *= corr; o3[r] *= corr;
    ptile[(r + hi) * 32 + col]      = (_Float16)p0;
    ptile[(r + hi) * 32 + col + 16] = (_Float16)p1;
  }
}

// ---------------------------------------------------------------------------
// Stage 0: fp32 -> fp16 conversion
// ---------------------------------------------------------------------------
__global__ void k_f32_to_f16(const float* __restrict__ in,
                             _Float16* __restrict__ out, int n) {
  int i = blockIdx.x * blockDim.x + threadIdx.x;
  int stride = gridDim.x * blockDim.x;
  for (; i < n; i += stride) out[i] = (_Float16)in[i];
}

// ---------------------------------------------------------------------------
// Stage 1: fused QKV projection, 32(M)x16(N) per wave (A reused 3x, and each
// B-fragment reused across 2 M-tiles -> 6 WMMA per K-step).
//   q = (X @ Wq^T + bq) * D^-0.5 -> Qs[b,h,t,d]
//   k =  X @ Wk^T + bk           -> Ks[b,h,t,d]
//   v =  X @ Wv^T + bv           -> Vt[b,h,d,t]  (transposed for PV B-frags)
// ---------------------------------------------------------------------------
__global__ void __launch_bounds__(128) k_qkv(
    const _Float16* __restrict__ Xh,
    const _Float16* __restrict__ Wqh, const _Float16* __restrict__ Wkh,
    const _Float16* __restrict__ Wvh,
    const float* __restrict__ bq, const float* __restrict__ bk,
    const float* __restrict__ bv,
    _Float16* __restrict__ Qs, _Float16* __restrict__ Ks,
    _Float16* __restrict__ Vt) {
  int lane = threadIdx.x & 31;
  int wave = threadIdx.x >> 5;
  int n0 = blockIdx.x * 32;
  int e0 = (blockIdx.y * 4 + wave) * 16;

  v8f cq0 = {}, ck0 = {}, cv0 = {};
  v8f cq1 = {}, ck1 = {}, cv1 = {};
  for (int k0 = 0; k0 < E_; k0 += 32) {
    v16h a0  = load_a_frag_g(Xh,  E_, n0,      k0, lane);
    v16h a1  = load_a_frag_g(Xh,  E_, n0 + 16, k0, lane);
    v16h bqf = load_b_frag_g(Wqh, E_, e0, k0, lane);
    v16h bkf = load_b_frag_g(Wkh, E_, e0, k0, lane);
    v16h bvf = load_b_frag_g(Wvh, E_, e0, k0, lane);
    if (k0 + 32 < E_)  // global_prefetch_b8, near-temporal
      __builtin_prefetch(Xh + (size_t)(n0 + (lane & 15)) * E_ + k0 + 32, 0, 3);
    cq0 = wmma16(a0, bqf, cq0);  cq1 = wmma16(a1, bqf, cq1);
    ck0 = wmma16(a0, bkf, ck0);  ck1 = wmma16(a1, bkf, ck1);
    cv0 = wmma16(a0, bvf, cv0);  cv1 = wmma16(a1, bvf, cv1);
  }

  int e = e0 + (lane & 15);
  int h = e >> 6;           // e / D
  int d = e & (D_ - 1);     // e % D
  float bqe = bq[e], bke = bk[e], bve = bv[e];
  int hi = (lane & 16) ? 8 : 0;
#pragma unroll
  for (int mt = 0; mt < 2; ++mt) {
    v8f& cq = mt ? cq1 : cq0;
    v8f& ck = mt ? ck1 : ck0;
    v8f& cv = mt ? cv1 : cv0;
#pragma unroll
    for (int r = 0; r < 8; ++r) {
      int n = n0 + mt * 16 + r + hi;
      int t = n >> 1;       // n / B
      int b = n & 1;        // n % B
      int bh = b * H_ + h;
      Qs[((size_t)bh * T_ + t) * D_ + d] = (_Float16)((cq[r] + bqe) * 0.125f);
      Ks[((size_t)bh * T_ + t) * D_ + d] = (_Float16)(ck[r] + bke);
      Vt[((size_t)bh * D_ + d) * T_ + t] = (_Float16)(cv[r] + bve);
    }
  }
}

// ---------------------------------------------------------------------------
// Stage 2: flash attention. 1 wave = one (b,h) and one 32-row Q block
// (two 16-row M-tiles sharing every K/V B-fragment -> halves K/V traffic).
// Per 32-wide S block: 8 WMMA scores + 8 WMMA P@V.
// ---------------------------------------------------------------------------
__global__ void __launch_bounds__(128) k_attn(
    const _Float16* __restrict__ Qs, const _Float16* __restrict__ Ks,
    const _Float16* __restrict__ Vt, _Float16* __restrict__ Ah) {
  __shared__ _Float16 plds[4][32][32];   // per-wave 32x32 P tile (2 KB each)
  int lane = threadIdx.x & 31;
  int wave = threadIdx.x >> 5;
  int wid  = blockIdx.x * 4 + wave;      // 2048 waves total
  int qb = wid & 63;                     // 64 Q blocks of 32 rows
  int bh = wid >> 6;                     // 0..31  (bh = b*H + h)

  const _Float16* Qb = Qs + (size_t)bh * T_ * D_;
  const _Float16* Kb = Ks + (size_t)bh * T_ * D_;
  const _Float16* Vb = Vt + (size_t)bh * D_ * T_;
  _Float16* pt0 = &plds[wave][0][0];
  _Float16* pt1 = &plds[wave][16][0];

  int q0 = qb * 32;
  v16h a00 = load_a_frag_g(Qb, D_, q0,      0,  lane);  // Mtile0, d=0..31
  v16h a01 = load_a_frag_g(Qb, D_, q0,      32, lane);  // Mtile0, d=32..63
  v16h a10 = load_a_frag_g(Qb, D_, q0 + 16, 0,  lane);  // Mtile1
  v16h a11 = load_a_frag_g(Qb, D_, q0 + 16, 32, lane);

  float m0[8], l0[8], m1[8], l1[8];
#pragma unroll
  for (int r = 0; r < 8; ++r) {
    m0[r] = -3.0e38f; l0[r] = 0.0f;
    m1[r] = -3.0e38f; l1[r] = 0.0f;
  }
  v8f o00 = {}, o01 = {}, o02 = {}, o03 = {};   // Mtile0, d-tiles 0..3
  v8f o10 = {}, o11 = {}, o12 = {}, o13 = {};   // Mtile1

  for (int s0 = 0; s0 < T_; s0 += 32) {
    // ---- K B-fragments, shared by both M-tiles
    v16h bk0 = load_b_frag_g(Kb, D_, s0,      0,  lane);
    v16h bk1 = load_b_frag_g(Kb, D_, s0,      32, lane);
    v16h bk2 = load_b_frag_g(Kb, D_, s0 + 16, 0,  lane);
    v16h bk3 = load_b_frag_g(Kb, D_, s0 + 16, 32, lane);

    // ---- scores: 2 M-tiles x 2 S-tiles, K = 64
    v8f sc00 = {}, sc01 = {}, sc10 = {}, sc11 = {};
    sc00 = wmma16(a00, bk0, sc00);  sc00 = wmma16(a01, bk1, sc00);
    sc01 = wmma16(a00, bk2, sc01);  sc01 = wmma16(a01, bk3, sc01);
    sc10 = wmma16(a10, bk0, sc10);  sc10 = wmma16(a11, bk1, sc10);
    sc11 = wmma16(a10, bk2, sc11);  sc11 = wmma16(a11, bk3, sc11);

    // ---- online softmax per M-tile; P goes to LDS for C->A re-layout
    softmax_update(sc00, sc01, m0, l0, o00, o01, o02, o03, pt0, lane);
    softmax_update(sc10, sc11, m1, l1, o10, o11, o12, o13, pt1, lane);
    asm volatile("s_wait_dscnt 0" ::: "memory");

    v16h pa0 = load_a_frag_lds(pt0, lane);
    v16h pa1 = load_a_frag_lds(pt1, lane);

    // ---- O += P @ V : V B-fragments shared by both M-tiles
    v16h bv0 = load_b_frag_g(Vb, T_, 0,  s0, lane);
    v16h bv1 = load_b_frag_g(Vb, T_, 16, s0, lane);
    v16h bv2 = load_b_frag_g(Vb, T_, 32, s0, lane);
    v16h bv3 = load_b_frag_g(Vb, T_, 48, s0, lane);
    o00 = wmma16(pa0, bv0, o00);  o10 = wmma16(pa1, bv0, o10);
    o01 = wmma16(pa0, bv1, o01);  o11 = wmma16(pa1, bv1, o11);
    o02 = wmma16(pa0, bv2, o02);  o12 = wmma16(pa1, bv2, o12);
    o03 = wmma16(pa0, bv3, o03);  o13 = wmma16(pa1, bv3, o13);
  }

  // ---- normalize and scatter to Ah [T*B, E] (f16) for the output GEMM
  int h = bh & (H_ - 1);
  int b = bh >> 4;
  int hi = (lane & 16) ? 8 : 0;
  int ecol = h * D_ + (lane & 15);
#pragma unroll
  for (int mt = 0; mt < 2; ++mt) {
    float* lr = mt ? l1 : l0;
    v8f& q0f = mt ? o10 : o00;
    v8f& q1f = mt ? o11 : o01;
    v8f& q2f = mt ? o12 : o02;
    v8f& q3f = mt ? o13 : o03;
#pragma unroll
    for (int r = 0; r < 8; ++r) {
      float inv = 1.0f / lr[r];
      int t = q0 + mt * 16 + r + hi;
      size_t nrow = (size_t)(t * B_ + b) * E_;
      Ah[nrow + ecol]      = (_Float16)(q0f[r] * inv);
      Ah[nrow + ecol + 16] = (_Float16)(q1f[r] * inv);
      Ah[nrow + ecol + 32] = (_Float16)(q2f[r] * inv);
      Ah[nrow + ecol + 48] = (_Float16)(q3f[r] * inv);
    }
  }
}

// ---------------------------------------------------------------------------
// Stage 3: output projection  out = Ah @ Wo^T + bo, 32x32 per wave
// (2x reuse on both A and B fragments, 4 WMMA per K-step).
// ---------------------------------------------------------------------------
__global__ void __launch_bounds__(128) k_out(
    const _Float16* __restrict__ Ah, const _Float16* __restrict__ Woh,
    const float* __restrict__ bo, float* __restrict__ out) {
  int lane = threadIdx.x & 31;
  int wave = threadIdx.x >> 5;
  int n0 = blockIdx.x * 32;
  int e0 = (blockIdx.y * 4 + wave) * 32;

  v8f c00 = {}, c01 = {}, c10 = {}, c11 = {};
  for (int k0 = 0; k0 < E_; k0 += 32) {
    v16h a0 = load_a_frag_g(Ah,  E_, n0,      k0, lane);
    v16h a1 = load_a_frag_g(Ah,  E_, n0 + 16, k0, lane);
    v16h b0 = load_b_frag_g(Woh, E_, e0,      k0, lane);
    v16h b1 = load_b_frag_g(Woh, E_, e0 + 16, k0, lane);
    if (k0 + 32 < E_)
      __builtin_prefetch(Ah + (size_t)(n0 + (lane & 15)) * E_ + k0 + 32, 0, 3);
    c00 = wmma16(a0, b0, c00);  c01 = wmma16(a0, b1, c01);
    c10 = wmma16(a1, b0, c10);  c11 = wmma16(a1, b1, c11);
  }
  int hi = (lane & 16) ? 8 : 0;
#pragma unroll
  for (int nt = 0; nt < 2; ++nt) {
#pragma unroll
    for (int et = 0; et < 2; ++et) {
      v8f& c = nt ? (et ? c11 : c10) : (et ? c01 : c00);
      int e = e0 + et * 16 + (lane & 15);
      float be = bo[e];
#pragma unroll
      for (int r = 0; r < 8; ++r) {
        int n = n0 + nt * 16 + r + hi;
        out[(size_t)n * E_ + e] = c[r] + be;
      }
    }
  }
}

// ---------------------------------------------------------------------------
// Host launcher
// ---------------------------------------------------------------------------
extern "C" void kernel_launch(void* const* d_in, const int* in_sizes, int n_in,
                              void* d_out, int out_size, void* d_ws, size_t ws_size,
                              hipStream_t stream) {
  const float* query = (const float*)d_in[0];
  const float* Wq = (const float*)d_in[1];
  const float* bq = (const float*)d_in[2];
  const float* Wk = (const float*)d_in[3];
  const float* bk = (const float*)d_in[4];
  const float* Wv = (const float*)d_in[5];
  const float* bv = (const float*)d_in[6];
  const float* Wo = (const float*)d_in[7];
  const float* bo = (const float*)d_in[8];
  float* out = (float*)d_out;

  // workspace layout (f16): X(4M) W*(4x1M) Q(4M) K(4M) Vt(4M) Ah(4M) = 48 MB
  _Float16* Xh  = (_Float16*)d_ws;
  _Float16* Wqh = Xh  + (size_t)N_ * E_;
  _Float16* Wkh = Wqh + (size_t)E_ * E_;
  _Float16* Wvh = Wkh + (size_t)E_ * E_;
  _Float16* Woh = Wvh + (size_t)E_ * E_;
  _Float16* Qs  = Woh + (size_t)E_ * E_;
  _Float16* Ks  = Qs  + (size_t)N_ * E_;   // B*H*T*D == N*E
  _Float16* Vt  = Ks  + (size_t)N_ * E_;
  _Float16* Ah  = Vt  + (size_t)N_ * E_;

  k_f32_to_f16<<<1024, 256, 0, stream>>>(query, Xh, N_ * E_);
  k_f32_to_f16<<<512,  256, 0, stream>>>(Wq, Wqh, E_ * E_);
  k_f32_to_f16<<<512,  256, 0, stream>>>(Wk, Wkh, E_ * E_);
  k_f32_to_f16<<<512,  256, 0, stream>>>(Wv, Wvh, E_ * E_);
  k_f32_to_f16<<<512,  256, 0, stream>>>(Wo, Woh, E_ * E_);

  // QKV: M-tiles of 32 (128 blocks) x 64 e-tiles (16 blocks of 4 waves)
  k_qkv<<<dim3(N_ / 32, E_ / 16 / 4), 128, 0, stream>>>(
      Xh, Wqh, Wkh, Wvh, bq, bk, bv, Qs, Ks, Vt);

  // 32 (b,h) * 64 Q-blocks = 2048 waves, 4 waves/block
  k_attn<<<dim3(512), 128, 0, stream>>>(Qs, Ks, Vt, Ah);

  // out: 128 x (8 blocks of 4 waves covering 32-wide e-tiles)
  k_out<<<dim3(N_ / 32, E_ / 32 / 4), 128, 0, stream>>>(Ah, Woh, bo, out);
}
